// MultiHeadAttention_38920993636954
// MI455X (gfx1250) — matrix operand order repaired and checked by reading, not compiled, not run on hardware
//
#include <hip/hip_runtime.h>
#include <stdint.h>

// ---------------------------------------------------------------------------
// MHA for MI455X (gfx1250), wave32, WMMA bf16 16x16x32, f32 accumulate.
// Pipeline:
//   cvt x->bf16, transpose+cvt weights ->bf16   (one-time, bandwidth-trivial)
//   GEMM  xb @ wqkvT -> qkv bf16      (async global->LDS double-buffered)
//   flash attention (TDM K-tile load, WMMA QK^T and PV, online softmax)
//   GEMM  att @ woutT -> out f32
// Workspace (48 MB): qkv | att | xb | wqkvT | woutT   (all bf16)
// ---------------------------------------------------------------------------

typedef __attribute__((ext_vector_type(16))) __bf16 v16bf;
typedef __attribute__((ext_vector_type(8)))  float  v8f;
typedef __attribute__((ext_vector_type(4)))  unsigned int v4u;
typedef __attribute__((ext_vector_type(8)))  int    v8i;
typedef __attribute__((ext_vector_type(4)))  int    v4i;

__device__ __forceinline__ unsigned short f2bf(float f) {
  union { float f; unsigned u; } x; x.f = f;
  unsigned r = x.u + 0x7FFFu + ((x.u >> 16) & 1u);   // round-to-nearest-even
  return (unsigned short)(r >> 16);
}

// Low 32 bits of a generic pointer to __shared__ = LDS byte address.
__device__ __forceinline__ unsigned lds_off(const void* p) {
  return (unsigned)(unsigned long long)p;
}

// Async global->LDS 16B copy per lane (ASYNCcnt-tracked, bypasses VGPRs).
__device__ __forceinline__ void async_copy_b128(unsigned lds_addr,
                                                const void* gptr) {
  asm volatile("global_load_async_to_lds_b128 %0, %1, off"
               :: "v"(lds_addr), "v"((unsigned long long)gptr)
               : "memory");
}
__device__ __forceinline__ void wait_async0() {
  asm volatile("s_wait_asynccnt 0" ::: "memory");
}

// A-fragment (16x32 bf16): lane holds row (lane&15); 16 contiguous K
// values {half*16 .. half*16+15} (same order as B-fragment).
__device__ __forceinline__ v16bf load_afrag(const unsigned short* p0) {
  union { v16bf v; uint4 q[2]; } u;
  u.q[0] = *reinterpret_cast<const uint4*>(p0);
  u.q[1] = *reinterpret_cast<const uint4*>(p0 + 8);
  return u.v;
}
// B-fragment (32x16 bf16): lane holds column (lane&15); 16 contiguous K
// values {half*16 .. half*16+15}.
__device__ __forceinline__ v16bf load_bfrag(const unsigned short* p0) {
  union { v16bf v; uint4 q[2]; } u;
  u.q[0] = *reinterpret_cast<const uint4*>(p0);
  u.q[1] = *reinterpret_cast<const uint4*>(p0 + 8);
  return u.v;
}

__device__ __forceinline__ v8f wmma_bf16(v16bf a, v16bf b, v8f c) {
  return __builtin_amdgcn_wmma_f32_16x16x32_bf16(
      false, a, false, b, (short)0, c, false, false);
}

// ---------------------------------------------------------------------------
// One-time conversion kernels
// ---------------------------------------------------------------------------
__global__ __launch_bounds__(256) void cvt_f32_bf16(
    const float* __restrict__ in, unsigned short* __restrict__ out) {
  const size_t i = ((size_t)blockIdx.x * blockDim.x + threadIdx.x) * 4;
  float4 v = *(const float4*)(in + i);
  uint2 pk;
  pk.x = (unsigned)f2bf(v.x) | ((unsigned)f2bf(v.y) << 16);
  pk.y = (unsigned)f2bf(v.z) | ((unsigned)f2bf(v.w) << 16);
  *(uint2*)(out + i) = pk;
}

// W [R,C] f32 -> Wt [C,R] bf16, LDS-tiled so both sides stay coalesced.
__global__ __launch_bounds__(256) void transpose_f32_bf16(
    const float* __restrict__ W, unsigned short* __restrict__ Wt,
    int R, int C) {
  __shared__ float tile[32][33];
  const int tx = threadIdx.x & 31, ty = threadIdx.x >> 5;  // 32x8
  const int c0 = blockIdx.x * 32, r0 = blockIdx.y * 32;
#pragma unroll
  for (int i = 0; i < 32; i += 8)
    tile[ty + i][tx] = W[(size_t)(r0 + ty + i) * C + c0 + tx];
  __syncthreads();
#pragma unroll
  for (int i = 0; i < 32; i += 8)
    Wt[(size_t)(c0 + ty + i) * R + r0 + tx] = f2bf(tile[tx][ty + i]);
}

// ---------------------------------------------------------------------------
// GEMM: C[M,N] = A[M,K] @ B[K,N], with A bf16 row-major and Bt = B^T bf16
// row-major [N,K]. 128x128 block tile, K-step 32, 8 waves (4x2), each 32x64.
// Double-buffered LDS fed by async global->LDS b128 copies.
// ---------------------------------------------------------------------------
template <bool C_BF16>
__global__ __launch_bounds__(256) void gemm_wmma_bf16(
    const unsigned short* __restrict__ A, const unsigned short* __restrict__ Bt,
    void* __restrict__ Cp, int M, int N, int K) {
  constexpr int BM = 128, BK = 32;
  constexpr int STR = 40;  // halves; 80B rows keep 16B alignment, rotate banks
  __shared__ __align__(16) unsigned short Asm[2][BM * STR];
  __shared__ __align__(16) unsigned short Bsm[2][BM * STR];

  const int tid  = threadIdx.x;
  const int lane = tid & 31, w = tid >> 5;
  const int half = lane >> 4, l15 = lane & 15;
  const int wr = w >> 1, wc = w & 1;
  const int m0 = blockIdx.y * BM, n0 = blockIdx.x * BM;

  auto stage = [&](int buf, int k0) {
    // 128 rows x 64B per tile = 512 b128 chunks; 2 per thread per matrix.
#pragma unroll
    for (int it = 0; it < 2; ++it) {
      const int idx = it * 256 + tid;
      const int r = idx >> 2, ch = idx & 3;  // row, 16B chunk in row
      async_copy_b128(lds_off(&Asm[buf][r * STR]) + ch * 16,
                      A + (size_t)(m0 + r) * K + k0 + ch * 8);
      async_copy_b128(lds_off(&Bsm[buf][r * STR]) + ch * 16,
                      Bt + (size_t)(n0 + r) * K + k0 + ch * 8);
    }
  };

  const v8f vzero = {0.f, 0.f, 0.f, 0.f, 0.f, 0.f, 0.f, 0.f};
  v8f acc[2][4];
#pragma unroll
  for (int mi = 0; mi < 2; ++mi)
#pragma unroll
    for (int ni = 0; ni < 4; ++ni) acc[mi][ni] = vzero;

  stage(0, 0);
  int buf = 0;
  for (int k0 = 0; k0 < K; k0 += BK) {
    wait_async0();        // own async copies for this tile done
    __syncthreads();      // everyone's copies done; prev compute done
    if (k0 + BK < K) stage(buf ^ 1, k0 + BK);  // prefetch next tile

    v16bf a[2], b[4];
#pragma unroll
    for (int mi = 0; mi < 2; ++mi)
      a[mi] = load_afrag(&Asm[buf][(wr * 32 + mi * 16 + l15) * STR + half * 16]);
#pragma unroll
    for (int ni = 0; ni < 4; ++ni)
      b[ni] = load_bfrag(&Bsm[buf][(wc * 64 + ni * 16 + l15) * STR + half * 16]);
#pragma unroll
    for (int mi = 0; mi < 2; ++mi)
#pragma unroll
      for (int ni = 0; ni < 4; ++ni)
        acc[mi][ni] = wmma_bf16(a[mi], b[ni], acc[mi][ni]);
    buf ^= 1;
  }

#pragma unroll
  for (int mi = 0; mi < 2; ++mi)
#pragma unroll
    for (int ni = 0; ni < 4; ++ni) {
      const int row0 = m0 + wr * 32 + mi * 16 + half * 8;
      const int col  = n0 + wc * 64 + ni * 16 + l15;
#pragma unroll
      for (int j = 0; j < 8; ++j) {
        const float v = acc[mi][ni][j];
        const size_t idx = (size_t)(row0 + j) * N + col;
        if (C_BF16) ((unsigned short*)Cp)[idx] = f2bf(v);
        else        ((float*)Cp)[idx] = v;
      }
    }
}

// ---------------------------------------------------------------------------
// Flash attention, causal. Workgroup = (b,h, 128-row q tile), 8 waves,
// wave w owns q rows [w*16, w*16+16). kv tiles of 64 keys.
// K tile loaded by the Tensor Data Mover (2D descriptor, HW row padding).
// ---------------------------------------------------------------------------
__global__ __launch_bounds__(256) void attn_flash_wmma(
    const unsigned short* __restrict__ qkv, unsigned short* __restrict__ att,
    int T) {
  constexpr int QT = 128, KT = 64, HD = 64, QKVD = 3072, D = 1024;
  constexpr int KSTR = 72, VSTR = 72, PSTR = 72;  // 144B rows, 16B-aligned
  __shared__ __align__(16) unsigned short Ksm[KT * KSTR];      // [key][d]
  __shared__ __align__(16) unsigned short Vsm[HD * VSTR];      // [d][key]
  __shared__ __align__(16) unsigned short Psm[8 * 16 * PSTR];  // per-wave P

  const int tid = threadIdx.x, lane = tid & 31, w = tid >> 5;
  const int half = lane >> 4, l15 = lane & 15;
  const int qt = blockIdx.x, bh = blockIdx.y;
  const int b = bh >> 4, h = bh & 15;
  const int q0 = qt * QT;
  const size_t rowB = (size_t)b * T;

  // Q fragments straight from global (bf16 row-major matches A-frag chunks)
  v16bf qf[2];
  {
    const int qrow = q0 + w * 16 + l15;
    const unsigned short* qp = qkv + (rowB + qrow) * QKVD + h * HD;
#pragma unroll
    for (int ks = 0; ks < 2; ++ks) qf[ks] = load_afrag(qp + ks * 32 + half * 16);
  }

  const v8f vzero = {0.f, 0.f, 0.f, 0.f, 0.f, 0.f, 0.f, 0.f};
  v8f o[4];
#pragma unroll
  for (int nt = 0; nt < 4; ++nt) o[nt] = vzero;
  float mrow[8], lrow[8];
#pragma unroll
  for (int j = 0; j < 8; ++j) { mrow[j] = -__builtin_inff(); lrow[j] = 0.f; }

  const int nkv = 2 * qt + 2;  // causal
  for (int kv = 0; kv < nkv; ++kv) {
    const int kv0 = kv * KT;
    __syncthreads();  // previous tile's consumers done

    // ---- K tile via TDM: 64x64 bf16, row stride 3072, LDS rows padded to
    //      72 halves via pad_interval=32DW (code 4), pad_amount=4DW (code 3).
    if (w == 0) {
      const unsigned long long ga = (unsigned long long)(const void*)
          (qkv + (rowB + kv0) * QKVD + D + h * HD);
      v4u g0;
      g0.x = 1u;                                     // count=1, user mode
      g0.y = lds_off(Ksm);                           // LDS byte address
      g0.z = (unsigned)ga;                           // global addr [31:0]
      g0.w = (unsigned)(ga >> 32) | (2u << 30);      // addr[56:32] | type=2
      v8i g1;
      g1[0] = (3 << 25) | (4 << 22) | (1 << 20) | (1 << 16);  // pad, 2B elems
      g1[1] = (int)(64u << 16);   // tensor_dim0[15:0] in [31:16]
      g1[2] = (int)(64u << 16);   // tensor_dim1[15:0] in [31:16]
      g1[3] = (int)(64u << 16);   // tile_dim0 in [31:16]
      g1[4] = 64;                 // tile_dim1 (tile_dim2=0)
      g1[5] = 3072;               // tensor_dim0_stride (elements)
      g1[6] = 0;
      g1[7] = 0;
      v4i gz = {0, 0, 0, 0};
#if defined(__clang_major__) && __clang_major__ >= 23
      v8i z8 = {0, 0, 0, 0, 0, 0, 0, 0};
      __builtin_amdgcn_tensor_load_to_lds(g0, g1, gz, gz, z8, 0);
#else
      __builtin_amdgcn_tensor_load_to_lds(g0, g1, gz, gz, 0);
#endif
      __builtin_amdgcn_s_wait_tensorcnt(0);
    }

    // ---- V tile transposed -> Vsm[d][key] (manual; TDM can't transpose)
    {
      int key = tid >> 4, d4 = (tid & 15) * 4;
#pragma unroll
      for (int it = 0; it < 4; ++it, key += 16) {
        uint2 dd = *(const uint2*)(qkv + (rowB + kv0 + key) * QKVD + 2 * D +
                                   h * HD + d4);
        Vsm[(d4 + 0) * VSTR + key] = (unsigned short)(dd.x & 0xFFFFu);
        Vsm[(d4 + 1) * VSTR + key] = (unsigned short)(dd.x >> 16);
        Vsm[(d4 + 2) * VSTR + key] = (unsigned short)(dd.y & 0xFFFFu);
        Vsm[(d4 + 3) * VSTR + key] = (unsigned short)(dd.y >> 16);
      }
    }
    __syncthreads();  // K (TDM) + V staged

    // ---- S = Q @ K^T
    v8f s[4];
#pragma unroll
    for (int nt = 0; nt < 4; ++nt) {
      v8f c = vzero;
#pragma unroll
      for (int ks = 0; ks < 2; ++ks) {
        v16bf bf = load_bfrag(Ksm + (nt * 16 + l15) * KSTR + ks * 32 + half * 16);
        c = wmma_bf16(qf[ks], bf, c);
      }
      s[nt] = c;
    }

    // ---- online softmax (row r at vgpr j=r&7 across one 16-lane half)
    const int qrow_base = q0 + w * 16 + 8 * half;
#pragma unroll
    for (int j = 0; j < 8; ++j) {
      const int qg = qrow_base + j;
      float sv[4];
      float rmax = -__builtin_inff();
#pragma unroll
      for (int nt = 0; nt < 4; ++nt) {
        const int kg = kv0 + nt * 16 + l15;
        float v = s[nt][j] * 0.125f;             // 1/sqrt(64)
        v = (kg <= qg) ? v : -__builtin_inff();  // causal mask
        sv[nt] = v;
        rmax = fmaxf(rmax, v);
      }
#pragma unroll
      for (int off = 1; off < 16; off <<= 1)
        rmax = fmaxf(rmax, __shfl_xor(rmax, off, 32));
      const float mnew  = fmaxf(mrow[j], rmax);
      const float scale = __expf(mrow[j] - mnew);
      float rsum = 0.f;
#pragma unroll
      for (int nt = 0; nt < 4; ++nt) {
        const float p = __expf(sv[nt] - mnew);
        sv[nt] = p;
        rsum += p;
      }
#pragma unroll
      for (int off = 1; off < 16; off <<= 1) rsum += __shfl_xor(rsum, off, 32);
      lrow[j] = lrow[j] * scale + rsum;
      mrow[j] = mnew;
#pragma unroll
      for (int nt = 0; nt < 4; ++nt) {
        o[nt][j] *= scale;
        Psm[(w * 16 + j + 8 * half) * PSTR + nt * 16 + l15] = f2bf(sv[nt]);
      }
    }

    // ---- O += P @ V (per-wave LDS relayout C-frag -> A-frag; V^T in LDS)
#pragma unroll
    for (int ks = 0; ks < 2; ++ks) {
      v16bf af = load_afrag(Psm + (w * 16 + l15) * PSTR + ks * 32 + half * 16);
#pragma unroll
      for (int nt = 0; nt < 4; ++nt) {
        v16bf bf = load_bfrag(Vsm + (nt * 16 + l15) * VSTR + ks * 32 + half * 16);
        o[nt] = wmma_bf16(af, bf, o[nt]);
      }
    }
  }

  // normalize and write attention output (bf16 [B*T, D])
#pragma unroll
  for (int nt = 0; nt < 4; ++nt)
#pragma unroll
    for (int j = 0; j < 8; ++j) {
      const int qg = q0 + w * 16 + 8 * half + j;
      const float invl = 1.0f / lrow[j];
      att[(rowB + qg) * D + h * HD + nt * 16 + l15] = f2bf(o[nt][j] * invl);
    }
}

// ---------------------------------------------------------------------------
// Launch. Inputs: {x f32, mask i32 (unused; causal applied analytically),
//                  w_qkv f32 [D,3D], w_out f32 [D,D]} -> out f32 [B,T,D].
// ---------------------------------------------------------------------------
extern "C" void kernel_launch(void* const* d_in, const int* in_sizes, int n_in,
                              void* d_out, int out_size, void* d_ws,
                              size_t ws_size, hipStream_t stream) {
  (void)in_sizes; (void)n_in; (void)out_size; (void)ws_size;
  const int B = 4, T = 1024, D = 1024;
  const float* x     = (const float*)d_in[0];
  const float* w_qkv = (const float*)d_in[2];
  const float* w_out = (const float*)d_in[3];
  float* out = (float*)d_out;

  unsigned short* qkv   = (unsigned short*)d_ws;               // [4096,3072]
  unsigned short* att   = qkv   + (size_t)B * T * 3 * D;       // [4096,1024]
  unsigned short* xb    = att   + (size_t)B * T * D;           // [4096,1024]
  unsigned short* wqkvT = xb    + (size_t)B * T * D;           // [3072,1024]
  unsigned short* woutT = wqkvT + (size_t)3 * D * D;           // [1024,1024]

  dim3 blk(256);
  // one-time casts/transposes (bandwidth-trivial vs the GEMM work)
  cvt_f32_bf16<<<dim3((B * T * D) / (256 * 4)), blk, 0, stream>>>(x, xb);
  transpose_f32_bf16<<<dim3(3 * D / 32, D / 32), blk, 0, stream>>>(
      w_qkv, wqkvT, D, 3 * D);
  transpose_f32_bf16<<<dim3(D / 32, D / 32), blk, 0, stream>>>(
      w_out, woutT, D, D);

  // xb[4096,1024] @ w_qkv -> qkv (bf16)
  gemm_wmma_bf16<true><<<dim3(3 * D / 128, B * T / 128), blk, 0, stream>>>(
      xb, wqkvT, qkv, B * T, 3 * D, D);
  // flash attention per (b,h) over 128-row q tiles
  attn_flash_wmma<<<dim3(T / 128, B * 16), blk, 0, stream>>>(qkv, att, T);
  // att[4096,1024] @ w_out -> out (f32)
  gemm_wmma_bf16<false><<<dim3(D / 128, B * T / 128), blk, 0, stream>>>(
      att, woutT, out, B * T, D, D);
}
